// NetVLAD_6270652252786
// MI455X (gfx1250) — matrix operand order repaired and checked by reading, not compiled, not run on hardware
//
#include <hip/hip_runtime.h>
#include <stdint.h>

// ---------------------------------------------------------------------------
// NetVLAD on MI455X (gfx1250).
// Strategy (memory-bound once GEMM runs on bf16 WMMA):
//   K0: enc_w fp32 -> bf16 (1 MB)
//   K1: per-row L2 normalize of x + fp32->bf16, staged through LDS with
//       global_load_async_to_lds_b128 (ASYNCcnt path). xn_bf16 = 128 MB in ws
//       (L2 is 192 MB -> GEMM A re-reads mostly hit L2).
//   K2: xc = xn @ enc_w^T + enc_b via v_wmma_f32_16x16x32_bf16, 128x128 block
//       tiles, result written straight into d_out[:, 9:, :] (no extra pass).
//   K3: logits (K=9) + softmax -> a (one thread per (n,t) row, no reductions)
//   K4: vlad[n,k,c] = sum_t a*xc - (sum_t a)*centroid, intra-norm over C,
//       written into d_out[:, 0:9, :].
// ---------------------------------------------------------------------------

#define EPSF 1e-12f

typedef __attribute__((ext_vector_type(16))) __bf16        v16bf;
typedef __attribute__((ext_vector_type(8)))  float         v8f;
typedef __attribute__((ext_vector_type(4)))  unsigned int  v4u;

union FragBF { v4u u[2]; v16bf b; };

__device__ __forceinline__ unsigned short f2bf_rne(float f) {
    // round-to-nearest-even fp32 -> bf16 (NaN not special-cased; inputs finite)
    unsigned u = __float_as_uint(f);
    unsigned r = u + 0x7FFFu + ((u >> 16) & 1u);
    return (unsigned short)(r >> 16);
}

// ---------------------------------------------------------------- K0: weights
__global__ void __launch_bounds__(256) cvt_w_kernel(
    const float* __restrict__ w, unsigned int* __restrict__ wbf, int n2)
{
    int i = blockIdx.x * 256 + threadIdx.x;
    if (i < n2) {
        float2 f = ((const float2*)w)[i];
        wbf[i] = (unsigned)f2bf_rne(f.x) | ((unsigned)f2bf_rne(f.y) << 16);
    }
}

// ------------------------------------------------- K1: normalize rows of x
// one block = one 1024-float row; row staged to LDS via async load-to-LDS
__global__ void __launch_bounds__(256) norm_rows_kernel(
    const float* __restrict__ x, unsigned short* __restrict__ xnbf)
{
    __shared__ float srow[1024];
    __shared__ float red[256];
    const int tid = threadIdx.x;
    const size_t rowbase = (size_t)blockIdx.x * 1024;

    const float* gsrc = x + rowbase + tid * 4;
    unsigned ldsaddr = (unsigned)(size_t)(&srow[tid * 4]);
    asm volatile("global_load_async_to_lds_b128 %0, %1, off"
                 :: "v"(ldsaddr), "v"(gsrc) : "memory");
    asm volatile("s_wait_asynccnt 0x0" ::: "memory");
    __syncthreads();

    float4 v = *(const float4*)(&srow[tid * 4]);
    red[tid] = v.x * v.x + v.y * v.y + v.z * v.z + v.w * v.w;
    __syncthreads();
    #pragma unroll
    for (int s = 128; s > 0; s >>= 1) {
        if (tid < s) red[tid] += red[tid + s];
        __syncthreads();
    }
    const float scale = 1.0f / fmaxf(sqrtf(red[0]), EPSF);

    uint2 o;
    o.x = (unsigned)f2bf_rne(v.x * scale) | ((unsigned)f2bf_rne(v.y * scale) << 16);
    o.y = (unsigned)f2bf_rne(v.z * scale) | ((unsigned)f2bf_rne(v.w * scale) << 16);
    *(uint2*)(xnbf + rowbase + tid * 4) = o;
}

// --------------------------------------------------------- K2: WMMA GEMM
// xc[r, c] = sum_k xn_bf16[r,k] * enc_w_bf16[c,k] + enc_b[c]
// grid = (512, 4); block = 256 (8 waves); 128x128 tile per block
__global__ void __launch_bounds__(256) gemm_wmma_kernel(
    const unsigned short* __restrict__ A,   // xn bf16  [65536][1024]
    const unsigned short* __restrict__ B,   // enc_w bf16 [512][1024]
    const float* __restrict__ bias,         // enc_b [512]
    float* __restrict__ out)                // (32, 2057, 512), xc at rows 9..
{
    const int lane = threadIdx.x & 31;
    const int wid  = threadIdx.x >> 5;
    const int mw   = wid >> 1;              // 0..3
    const int nw   = wid & 1;               // 0..1
    const int l15  = lane & 15;
    const int hi   = lane >> 4;             // 0 or 1
    const int Mbase = blockIdx.x * 128 + mw * 32;
    const int Nbase = blockIdx.y * 128 + nw * 64;

    const unsigned short* arow0p = A + (size_t)(Mbase + l15) * 1024;
    const unsigned short* arow1p = arow0p + 16 * 1024;
    const unsigned short* bp[4];
    #pragma unroll
    for (int nt = 0; nt < 4; ++nt)
        bp[nt] = B + (size_t)(Nbase + nt * 16 + l15) * 1024;

    v8f zero = {};
    v8f acc[2][4];
    #pragma unroll
    for (int mt = 0; mt < 2; ++mt)
        #pragma unroll
        for (int nt = 0; nt < 4; ++nt) acc[mt][nt] = zero;

    for (int k0 = 0; k0 < 1024; k0 += 32) {
        FragBF fa[2], fb[4];
        // A 16x32 bf16 frag: lanes<16: K [k0, k0+8) and [k0+16, k0+24); lanes>=16: +8
        const int ka = k0 + hi * 8;
        fa[0].u[0] = *(const v4u*)(arow0p + ka);
        fa[0].u[1] = *(const v4u*)(arow0p + ka + 16);
        fa[1].u[0] = *(const v4u*)(arow1p + ka);
        fa[1].u[1] = *(const v4u*)(arow1p + ka + 16);
        // B 32x16 bf16 frag: lane = col, K [k0, k0+8) and [k0+16, k0+24) (lanes>=16: +8)
        const int kb = k0 + hi * 8;
        #pragma unroll
        for (int nt = 0; nt < 4; ++nt) {
            fb[nt].u[0] = *(const v4u*)(bp[nt] + kb);
            fb[nt].u[1] = *(const v4u*)(bp[nt] + kb + 16);
        }
        #pragma unroll
        for (int mt = 0; mt < 2; ++mt)
            #pragma unroll
            for (int nt = 0; nt < 4; ++nt)
                acc[mt][nt] = __builtin_amdgcn_wmma_f32_16x16x32_bf16(
                    false, fa[mt].b, false, fb[nt].b,
                    (short)0, acc[mt][nt], false, false);
    }

    // epilogue: C/D layout -> VGPR j: M = j + 8*hi, N = l15
    #pragma unroll
    for (int mt = 0; mt < 2; ++mt) {
        #pragma unroll
        for (int nt = 0; nt < 4; ++nt) {
            const int col = Nbase + nt * 16 + l15;
            const float b = bias[col];
            #pragma unroll
            for (int j = 0; j < 8; ++j) {
                const int r = Mbase + mt * 16 + j + hi * 8;
                const int n = r >> 11;
                const int t = r & 2047;
                out[((size_t)n * 2057 + 9 + t) * 512 + col] = acc[mt][nt][j] + b;
            }
        }
    }
}

// ----------------------------------------------- K3: logits + softmax -> a
// one thread per (n,t) row; conv_w index is wave-uniform -> scalar loads
__global__ void __launch_bounds__(256) softassign_kernel(
    const float* __restrict__ out,          // xc lives in d_out rows 9..
    const float* __restrict__ conv_w,       // (9, 512)
    const float* __restrict__ conv_b,       // (9,)
    float* __restrict__ a)                  // (65536, 9)
{
    const int r = blockIdx.x * 256 + threadIdx.x;
    const int n = r >> 11;
    const int t = r & 2047;
    const float* xc = out + ((size_t)n * 2057 + 9 + t) * 512;

    float acc[9];
    #pragma unroll
    for (int k = 0; k < 9; ++k) acc[k] = conv_b[k];

    for (int c = 0; c < 512; c += 4) {
        float4 xv = *(const float4*)(xc + c);
        #pragma unroll
        for (int k = 0; k < 9; ++k) {
            const float* w = conv_w + k * 512 + c;
            acc[k] += xv.x * w[0] + xv.y * w[1] + xv.z * w[2] + xv.w * w[3];
        }
    }
    float m = acc[0];
    #pragma unroll
    for (int k = 1; k < 9; ++k) m = fmaxf(m, acc[k]);
    float s = 0.f;
    float e[9];
    #pragma unroll
    for (int k = 0; k < 9; ++k) { e[k] = __expf(acc[k] - m); s += e[k]; }
    const float inv = 1.0f / s;
    #pragma unroll
    for (int k = 0; k < 9; ++k) a[(size_t)r * 9 + k] = e[k] * inv;
}

// ------------------------------------------- K4: vlad + intra-norm -> out
// one block per n (512 threads, thread = channel c)
__global__ void __launch_bounds__(512) vlad_kernel(
    const float* __restrict__ a,            // (65536, 9)
    const float* __restrict__ cent,         // (9, 512)
    float* __restrict__ out)                // (32, 2057, 512)
{
    const int n = blockIdx.x;
    const int c = threadIdx.x;
    const int lane = c & 31, wid = c >> 5;

    float acc[9], asum[9];
    #pragma unroll
    for (int k = 0; k < 9; ++k) { acc[k] = 0.f; asum[k] = 0.f; }

    const float* arow = a + (size_t)n * 2048 * 9;
    const float* xcb  = out + ((size_t)n * 2057 + 9) * 512 + c;
    for (int t = 0; t < 2048; ++t) {
        const float xv = xcb[(size_t)t * 512];
        const float* ap = arow + t * 9;              // wave-uniform
        #pragma unroll
        for (int k = 0; k < 9; ++k) {
            const float av = ap[k];
            acc[k]  += av * xv;
            asum[k] += av;
        }
    }
    float vl[9];
    #pragma unroll
    for (int k = 0; k < 9; ++k) vl[k] = acc[k] - asum[k] * cent[k * 512 + c];

    __shared__ float sred[16];
    __shared__ float snorm[9];
    for (int k = 0; k < 9; ++k) {
        float v = vl[k] * vl[k];
        #pragma unroll
        for (int off = 16; off > 0; off >>= 1) v += __shfl_down(v, off, 32);
        __syncthreads();
        if (lane == 0) sred[wid] = v;
        __syncthreads();
        if (c == 0) {
            float s = 0.f;
            #pragma unroll
            for (int i = 0; i < 16; ++i) s += sred[i];
            snorm[k] = s;
        }
    }
    __syncthreads();
    #pragma unroll
    for (int k = 0; k < 9; ++k) {
        const float sc = 1.0f / fmaxf(sqrtf(snorm[k]), EPSF);
        out[((size_t)n * 2057 + k) * 512 + c] = vl[k] * sc;
    }
}

// ---------------------------------------------------------------------------
extern "C" void kernel_launch(void* const* d_in, const int* in_sizes, int n_in,
                              void* d_out, int out_size, void* d_ws, size_t ws_size,
                              hipStream_t stream)
{
    const float* x         = (const float*)d_in[0];  // (32,2048,1024)
    const float* centroids = (const float*)d_in[1];  // (9,512)
    const float* enc_w     = (const float*)d_in[2];  // (512,1024)
    const float* enc_b     = (const float*)d_in[3];  // (512,)
    const float* conv_w    = (const float*)d_in[4];  // (9,512)
    const float* conv_b    = (const float*)d_in[5];  // (9,)
    float* out = (float*)d_out;                      // (32,2057,512)

    char* ws = (char*)d_ws;
    unsigned short* xnbf = (unsigned short*)ws;                          // 128 MB
    unsigned short* wbf  = (unsigned short*)(ws + (size_t)134217728);    // 1 MB
    float*          avec = (float*)(ws + (size_t)134217728 + 1048576);   // 2.25 MB

    cvt_w_kernel<<<1024, 256, 0, stream>>>(enc_w, (unsigned int*)wbf, 262144);
    norm_rows_kernel<<<65536, 256, 0, stream>>>(x, xnbf);
    dim3 g2(512, 4, 1);
    gemm_wmma_kernel<<<g2, 256, 0, stream>>>(xnbf, wbf, enc_b, out);
    softassign_kernel<<<256, 256, 0, stream>>>(out, conv_w, conv_b, avec);
    vlad_kernel<<<32, 512, 0, stream>>>(avec, centroids, out);
}
